// MultiHeadAttentionLayer_3083786518922
// MI455X (gfx1250) — matrix operand order repaired and checked by reading, not compiled, not run on hardware
//
#include <hip/hip_runtime.h>
#include <math.h>

// ---------------------------------------------------------------------------
// MHA layer for MI455X (gfx1250): bf16 WMMA GEMMs + flash-style attention.
// B=4, S=2048, D=1024, H=16, HD=64.
// ---------------------------------------------------------------------------

#define Bb 4
#define Ss 2048
#define Dd 1024
#define Hh 16
#define HDh 64

typedef __attribute__((ext_vector_type(16))) __bf16 bf16x16;
typedef __attribute__((ext_vector_type(8)))  __bf16 bf16x8;
typedef __attribute__((ext_vector_type(8)))  float  f32x8;

union FragAB {
    bf16x16        v;
    bf16x8         h[2];
    unsigned short u[16];
};

union Pack8 {
    uint4          v;
    unsigned short u[8];
};

__device__ __forceinline__ unsigned short f2bf(float f) {
    unsigned int u = __float_as_uint(f);
    if ((u & 0x7fffffffu) > 0x7f800000u) return (unsigned short)((u >> 16) | 0x0040u); // NaN
    u += 0x7fffu + ((u >> 16) & 1u);   // round to nearest even
    return (unsigned short)(u >> 16);
}

__device__ __forceinline__ f32x8 wmma_bf16(bf16x16 a, bf16x16 b, f32x8 c) {
    // (neg_a, A, neg_b, B, c_mod, C, reuse_a, reuse_b)
    return __builtin_amdgcn_wmma_f32_16x16x32_bf16(false, a, false, b, (short)0, c,
                                                   false, false);
}

__device__ __forceinline__ f32x8 fzero8() {
    f32x8 z;
#pragma unroll
    for (int r = 0; r < 8; ++r) z[r] = 0.0f;
    return z;
}

// ---------------------------------------------------------------------------
// fp32 -> bf16, 8 elements per thread: 2x float4 loads + 1x b128 store
// ---------------------------------------------------------------------------
__global__ __launch_bounds__(256)
void cvt_bf16x8_kernel(const float* __restrict__ src,
                       unsigned short* __restrict__ dst, size_t n8) {
    const size_t i = (size_t)blockIdx.x * blockDim.x + threadIdx.x;
    if (i >= n8) return;
    const float4* s = (const float4*)src + 2 * i;
    const float4 a = s[0];
    const float4 b = s[1];
    uint4 o;
    o.x = (unsigned)f2bf(a.x) | ((unsigned)f2bf(a.y) << 16);
    o.y = (unsigned)f2bf(a.z) | ((unsigned)f2bf(a.w) << 16);
    o.z = (unsigned)f2bf(b.x) | ((unsigned)f2bf(b.y) << 16);
    o.w = (unsigned)f2bf(b.z) | ((unsigned)f2bf(b.w) << 16);
    ((uint4*)dst)[i] = o;
}

// ---------------------------------------------------------------------------
// C(M,N) = A(M,K) * Bw(N,K)^T   -- both operands K-major (NT GEMM)
// block = (32,4): 4 waves, each wave computes a 64x64 tile (4x4 WMMA frags).
// block tile = 128x128.  A frag: row per lane, K packed.  B frag: col per
// lane, K packed -> 16B contiguous global_load_b128 per lane for every frag.
// ---------------------------------------------------------------------------
template <bool OUT_F32>
__global__ __launch_bounds__(128)
void gemm_nt_kernel(const unsigned short* __restrict__ A,
                    const unsigned short* __restrict__ Bw,
                    void* __restrict__ Cout, int M, int N, int K) {
    const int lane = threadIdx.x;
    const int wave = threadIdx.y;
    const int lm   = lane & 15;
    const int hi   = lane >> 4;                     // 0 / 1
    const int m0   = blockIdx.y * 128 + (wave >> 1) * 64;
    const int n0   = blockIdx.x * 128 + (wave & 1) * 64;

    f32x8 acc[4][4];
#pragma unroll
    for (int i = 0; i < 4; ++i)
#pragma unroll
        for (int j = 0; j < 4; ++j) acc[i][j] = fzero8();

    for (int k0 = 0; k0 < K; k0 += 32) {
        FragAB a[4], b[4];
#pragma unroll
        for (int i = 0; i < 4; ++i) {
            // A 16x32 bf16 frag: lane holds row m, K chunks {hi*8..+7, 16+hi*8..+7}
            const unsigned short* pa =
                A + (size_t)(m0 + 16 * i + lm) * K + k0 + hi * 8;
            a[i].h[0] = *(const bf16x8*)pa;
            a[i].h[1] = *(const bf16x8*)(pa + 16);
        }
#pragma unroll
        for (int j = 0; j < 4; ++j) {
            // B 32x16 frag: lane holds col n (= row n of Bw), K chunks {hi*8..+7, 16+hi*8..+7}
            const unsigned short* pb =
                Bw + (size_t)(n0 + 16 * j + lm) * K + k0 + hi * 8;
            b[j].h[0] = *(const bf16x8*)pb;
            b[j].h[1] = *(const bf16x8*)(pb + 16);
        }
#pragma unroll
        for (int i = 0; i < 4; ++i)
#pragma unroll
            for (int j = 0; j < 4; ++j)
                acc[i][j] = wmma_bf16(a[i].v, b[j].v, acc[i][j]);
    }

#pragma unroll
    for (int i = 0; i < 4; ++i)
#pragma unroll
        for (int j = 0; j < 4; ++j)
#pragma unroll
            for (int r = 0; r < 8; ++r) {
                // C frag: VGPR r -> row r (lanes 0-15) / row r+8 (lanes 16-31)
                const int m = m0 + 16 * i + r + hi * 8;
                const int n = n0 + 16 * j + lm;
                const float val = acc[i][j][r];
                if (OUT_F32)
                    ((float*)Cout)[(size_t)m * N + n] = val;
                else
                    ((unsigned short*)Cout)[(size_t)m * N + n] = f2bf(val);
            }
}

// ---------------------------------------------------------------------------
// V (B,S,D) bf16 -> Vt (B,H,HD,S) bf16, LDS tile transpose.
// grid (S/64, B*H), block 256.  Coalesced b128 reads AND writes; the
// transpose scatter happens inside LDS (rows padded to 72 shorts so row
// stride = 144B and every 8-short chunk stays 16B-aligned).
// ---------------------------------------------------------------------------
__global__ __launch_bounds__(256)
void transpose_v_kernel(const unsigned short* __restrict__ Vp,
                        unsigned short* __restrict__ Vt) {
    __shared__ __align__(16) unsigned short tile[64][72];
    const int t   = threadIdx.x;
    const int bh  = blockIdx.y;
    const int b   = bh >> 4;
    const int h   = bh & 15;
    const int s0  = blockIdx.x * 64;
    const int c8  = t & 7;    // which 8-element chunk
    const int row = t >> 3;   // 0..31

    // phase 1: read 64 s-rows of Vp (b128, coalesced), scatter into tile[d][s]
#pragma unroll
    for (int p = 0; p < 2; ++p) {
        const int s = row + p * 32;
        Pack8 v;
        v.v = *(const uint4*)(Vp + (size_t)(b * Ss + s0 + s) * Dd + h * HDh + c8 * 8);
#pragma unroll
        for (int i = 0; i < 8; ++i) tile[c8 * 8 + i][s] = v.u[i];
    }
    __syncthreads();

    // phase 2: read tile rows (d fixed, s contiguous) and b128-store to Vt
#pragma unroll
    for (int p = 0; p < 2; ++p) {
        const int d = row + p * 32;
        const uint4 v = *(const uint4*)&tile[d][c8 * 8];
        *(uint4*)(Vt + ((size_t)(b * Hh + h) * HDh + d) * Ss + s0 + c8 * 8) = v;
    }
}

// ---------------------------------------------------------------------------
// Flash attention: grid (S/64, B*H), block (32,4).  Wave w owns the 16-query
// strip q0 = blockIdx.x*64 + w*16.  Streams over 64-key tiles: Q*K^T via
// WMMA, online softmax (row stats via width-16 shfl_xor), P re-laid out
// C-frag -> A-frag through wave-private LDS, then P*V via WMMA.
// ---------------------------------------------------------------------------
__global__ __launch_bounds__(128)
void flash_attn_kernel(const unsigned short* __restrict__ Qp,  // (B,S,D) bf16
                       const unsigned short* __restrict__ Kp,  // (B,S,D) bf16
                       const unsigned short* __restrict__ Vt,  // (B,H,HD,S) bf16
                       const int* __restrict__ mask,           // (B,1,S,S)
                       unsigned short* __restrict__ Y) {       // (B,S,D) bf16
    __shared__ __align__(16) unsigned short ldsP[4][16][72];

    const int lane = threadIdx.x;
    const int wave = threadIdx.y;
    const int lm   = lane & 15;
    const int hi   = lane >> 4;
    const int bh   = blockIdx.y;
    const int b    = bh >> 4;
    const int h    = bh & 15;
    const int q0   = blockIdx.x * 64 + wave * 16;
    const float scale = 0.125f;  // 1/sqrt(64)

    // Q A-fragments for this 16-row strip, resident all loop (HD=64 -> 2 frags)
    FragAB qa[2];
    {
        const size_t rowQ = (size_t)(b * Ss + q0 + lm) * Dd + h * HDh;
#pragma unroll
        for (int kk = 0; kk < 2; ++kk) {
            const unsigned short* p = Qp + rowQ + kk * 32 + hi * 8;
            qa[kk].h[0] = *(const bf16x8*)p;
            qa[kk].h[1] = *(const bf16x8*)(p + 16);
        }
    }

    f32x8 acc[4];
#pragma unroll
    for (int jd = 0; jd < 4; ++jd) acc[jd] = fzero8();
    float mrow[8], lrow[8];
#pragma unroll
    for (int r = 0; r < 8; ++r) { mrow[r] = -INFINITY; lrow[r] = 0.0f; }

    for (int kt0 = 0; kt0 < Ss; kt0 += 64) {
        // ---- scores: S = Q * K^T (16 x 64) -------------------------------
        f32x8 sc[4];
#pragma unroll
        for (int j = 0; j < 4; ++j) {
            const unsigned short* pk =
                Kp + (size_t)(b * Ss + kt0 + 16 * j + lm) * Dd + h * HDh + hi * 8;
            FragAB bk0, bk1;
            bk0.h[0] = *(const bf16x8*)pk;
            bk0.h[1] = *(const bf16x8*)(pk + 16);
            bk1.h[0] = *(const bf16x8*)(pk + 32);
            bk1.h[1] = *(const bf16x8*)(pk + 48);
            f32x8 s = fzero8();
            s = wmma_bf16(qa[0].v, bk0.v, s);
            s = wmma_bf16(qa[1].v, bk1.v, s);
            sc[j] = s;
        }

        // ---- scale + mask + per-block row max ----------------------------
        float bmax[8];
#pragma unroll
        for (int r = 0; r < 8; ++r) {
            const int q = q0 + r + hi * 8;
            const int* mp = mask + ((size_t)b * Ss + q) * Ss + kt0 + lm;
            float mx = -INFINITY;
#pragma unroll
            for (int j = 0; j < 4; ++j) {
                float v = sc[j][r] * scale;
                if (mp[j * 16] == 0) v = -INFINITY;
                sc[j][r] = v;
                mx = fmaxf(mx, v);
            }
            bmax[r] = mx;
        }
#pragma unroll
        for (int r = 0; r < 8; ++r)
#pragma unroll
            for (int off = 1; off < 16; off <<= 1)
                bmax[r] = fmaxf(bmax[r], __shfl_xor(bmax[r], off, 16));

        // ---- online softmax update ---------------------------------------
        float alpha[8];
#pragma unroll
        for (int r = 0; r < 8; ++r) {
            const float mnew = fmaxf(mrow[r], bmax[r]);
            alpha[r] = (mrow[r] == -INFINITY) ? 0.0f : __expf(mrow[r] - mnew);
            mrow[r] = mnew;
        }
        float bsum[8];
#pragma unroll
        for (int r = 0; r < 8; ++r) {
            float s = 0.0f;
#pragma unroll
            for (int j = 0; j < 4; ++j) {
                const float sv = sc[j][r];
                const float p = (sv == -INFINITY) ? 0.0f : __expf(sv - mrow[r]);
                sc[j][r] = p;
                s += p;
            }
            bsum[r] = s;
        }
#pragma unroll
        for (int r = 0; r < 8; ++r)
#pragma unroll
            for (int off = 1; off < 16; off <<= 1)
                bsum[r] += __shfl_xor(bsum[r], off, 16);
#pragma unroll
        for (int r = 0; r < 8; ++r) lrow[r] = lrow[r] * alpha[r] + bsum[r];
#pragma unroll
        for (int jd = 0; jd < 4; ++jd)
#pragma unroll
            for (int r = 0; r < 8; ++r) acc[jd][r] *= alpha[r];

        // ---- P: C-frag layout -> A-frag layout via wave-private LDS ------
#pragma unroll
        for (int r = 0; r < 8; ++r)
#pragma unroll
            for (int j = 0; j < 4; ++j)
                ldsP[wave][r + hi * 8][j * 16 + lm] = f2bf(sc[j][r]);
        // (same-wave LDS RAW: compiler inserts s_wait_dscnt)
        FragAB pa[2];
#pragma unroll
        for (int kk = 0; kk < 2; ++kk) {
            pa[kk].h[0] = *(const bf16x8*)&ldsP[wave][lm][kk * 32 + hi * 8];
            pa[kk].h[1] = *(const bf16x8*)&ldsP[wave][lm][kk * 32 + 16 + hi * 8];
        }

        // ---- acc += P * V -------------------------------------------------
#pragma unroll
        for (int kk = 0; kk < 2; ++kk)
#pragma unroll
            for (int jd = 0; jd < 4; ++jd) {
                const unsigned short* pv =
                    Vt + ((size_t)(b * Hh + h) * HDh + jd * 16 + lm) * Ss +
                    kt0 + kk * 32 + hi * 8;
                FragAB bv;
                bv.h[0] = *(const bf16x8*)pv;
                bv.h[1] = *(const bf16x8*)(pv + 16);
                acc[jd] = wmma_bf16(pa[kk].v, bv.v, acc[jd]);
            }
    }

    // ---- epilogue: y = acc / l -> Y (B,S,D) bf16 -------------------------
#pragma unroll
    for (int r = 0; r < 8; ++r) {
        const int q = q0 + r + hi * 8;
        const float inv = 1.0f / lrow[r];
#pragma unroll
        for (int jd = 0; jd < 4; ++jd)
            Y[(size_t)(b * Ss + q) * Dd + h * HDh + jd * 16 + lm] =
                f2bf(acc[jd][r] * inv);
    }
}

// ---------------------------------------------------------------------------
extern "C" void kernel_launch(void* const* d_in, const int* in_sizes, int n_in,
                              void* d_out, int out_size, void* d_ws, size_t ws_size,
                              hipStream_t stream) {
    const float* q    = (const float*)d_in[0];
    const float* k    = (const float*)d_in[1];
    const float* v    = (const float*)d_in[2];
    const int*   mask = (const int*)d_in[3];
    const float* Wq   = (const float*)d_in[4];
    const float* Wk   = (const float*)d_in[5];
    const float* Wv   = (const float*)d_in[6];
    const float* Wo   = (const float*)d_in[7];

    char* ws = (char*)d_ws;
    const size_t XB = (size_t)Bb * Ss * Dd * 2;  // 16 MiB bf16 activation buffer
    const size_t WB = (size_t)Dd * Dd * 2;       // 2 MiB bf16 weight buffer

    unsigned short* Xq   = (unsigned short*)(ws);
    unsigned short* Xk   = (unsigned short*)(ws + XB);
    unsigned short* Xv   = (unsigned short*)(ws + 2 * XB);
    unsigned short* Wq16 = (unsigned short*)(ws + 3 * XB);
    unsigned short* Wk16 = (unsigned short*)(ws + 3 * XB + WB);
    unsigned short* Wv16 = (unsigned short*)(ws + 3 * XB + 2 * WB);
    unsigned short* Wo16 = (unsigned short*)(ws + 3 * XB + 3 * WB);
    unsigned short* Qp   = (unsigned short*)(ws + 3 * XB + 4 * WB);
    unsigned short* Kp   = (unsigned short*)(ws + 4 * XB + 4 * WB);
    unsigned short* Vp   = (unsigned short*)(ws + 5 * XB + 4 * WB);
    unsigned short* Vt   = Xq;  // reuse: Xq consumed after Q projection
    unsigned short* Yb   = Xk;  // reuse: Xk consumed after K projection

    const size_t nx8 = (size_t)Bb * Ss * Dd / 8;  // 1048576
    const size_t nw8 = (size_t)Dd * Dd / 8;       // 131072
    const int M = Bb * Ss, N = Dd, K = Dd;

    // 1) fp32 -> bf16 staging (8 elems/thread, b128 stores)
    cvt_bf16x8_kernel<<<(unsigned)(nx8 / 256), 256, 0, stream>>>(q, Xq, nx8);
    cvt_bf16x8_kernel<<<(unsigned)(nx8 / 256), 256, 0, stream>>>(k, Xk, nx8);
    cvt_bf16x8_kernel<<<(unsigned)(nx8 / 256), 256, 0, stream>>>(v, Xv, nx8);
    cvt_bf16x8_kernel<<<(unsigned)(nw8 / 256), 256, 0, stream>>>(Wq, Wq16, nw8);
    cvt_bf16x8_kernel<<<(unsigned)(nw8 / 256), 256, 0, stream>>>(Wk, Wk16, nw8);
    cvt_bf16x8_kernel<<<(unsigned)(nw8 / 256), 256, 0, stream>>>(Wv, Wv16, nw8);
    cvt_bf16x8_kernel<<<(unsigned)(nw8 / 256), 256, 0, stream>>>(Wo, Wo16, nw8);

    // 2) Q/K/V projections (bf16 out)
    dim3 blk(32, 4);
    dim3 gproj(N / 128, M / 128);
    gemm_nt_kernel<false><<<gproj, blk, 0, stream>>>(Xq, Wq16, Qp, M, N, K);
    gemm_nt_kernel<false><<<gproj, blk, 0, stream>>>(Xk, Wk16, Kp, M, N, K);
    gemm_nt_kernel<false><<<gproj, blk, 0, stream>>>(Xv, Wv16, Vp, M, N, K);

    // 3) per-head transpose of V (LDS tile transpose, b128 both sides)
    dim3 gtr(Ss / 64, Bb * Hh);
    transpose_v_kernel<<<gtr, 256, 0, stream>>>(Vp, Vt);

    // 4) flash attention
    dim3 gatt(Ss / 64, Bb * Hh);
    flash_attn_kernel<<<gatt, blk, 0, stream>>>(Qp, Kp, Vt, mask, Yb);

    // 5) output projection (fp32 out)
    gemm_nt_kernel<true><<<gproj, blk, 0, stream>>>(Yb, Wo16, d_out, M, N, K);
}
